// LayerNormLSTMCell_4647154614506
// MI455X (gfx1250) — compile-verified, matrix operand order checked by reading, not run on hardware
//
#include <hip/hip_runtime.h>
#include <stdint.h>

// ---------------------------------------------------------------------------
// Types for the gfx1250 WMMA / async builtins
// ---------------------------------------------------------------------------
typedef __attribute__((ext_vector_type(16))) __bf16 v16bf;
typedef __attribute__((ext_vector_type(8)))  __bf16 v8bf;
typedef __attribute__((ext_vector_type(8)))  float  v8f;
typedef int v4i __attribute__((vector_size(16)));   // matches async-LDS builtin param
typedef unsigned short u16;

#define AS1 __attribute__((address_space(1)))
#define AS3 __attribute__((address_space(3)))

#if __has_builtin(__builtin_amdgcn_global_load_async_to_lds_b128)
#define HAVE_ASYNC_LDS 1
#else
#define HAVE_ASYNC_LDS 0
#endif

__device__ __forceinline__ void async_wait_le4() {
#if HAVE_ASYNC_LDS
#if __has_builtin(__builtin_amdgcn_s_wait_asynccnt)
  __builtin_amdgcn_s_wait_asynccnt(4);
#else
  asm volatile("s_wait_asynccnt 0x4" ::: "memory");
#endif
#endif
}
__device__ __forceinline__ void async_wait_le0() {
#if HAVE_ASYNC_LDS
#if __has_builtin(__builtin_amdgcn_s_wait_asynccnt)
  __builtin_amdgcn_s_wait_asynccnt(0);
#else
  asm volatile("s_wait_asynccnt 0x0" ::: "memory");
#endif
#endif
}

// ---------------------------------------------------------------------------
// bf16 helpers (bit-exact, no __bf16 arithmetic needed)
// ---------------------------------------------------------------------------
__device__ __forceinline__ u16 f32_to_bf16(float f) {
  unsigned u = __float_as_uint(f);
  unsigned r = u + 0x7fffu + ((u >> 16) & 1u);   // round-to-nearest-even
  return (u16)(r >> 16);
}
__device__ __forceinline__ float bf16_to_f32(u16 b) {
  return __uint_as_float(((unsigned)b) << 16);
}

// Split f32 -> bf16 hi + bf16 lo so that hi+lo ~= x (Markidis-style).
__global__ __launch_bounds__(256) void split_kernel(const float* __restrict__ src,
                                                    u16* __restrict__ hi,
                                                    u16* __restrict__ lo, int n) {
  int i = blockIdx.x * 256 + threadIdx.x;
  if (i < n) {
    float x = src[i];
    u16 h = f32_to_bf16(x);
    hi[i] = h;
    lo[i] = f32_to_bf16(x - bf16_to_f32(h));
  }
}

// ---------------------------------------------------------------------------
// GEMM: C[M,N] = X[M,K] * W[N,K]^T  (bf16x3 split products).
//
// Block = 256 thr (8 wave32), tile 128(M) x 128(N), Kstep 32, waves 4M x 2N.
// Per wave: 32x64 (2 M-subtiles x 4 N-subtiles) -> 24 WMMA / K-step.
// W tile (128 rows x 32 K x {hi,lo}) is staged in LDS double-buffered via
// CDNA5 async global->LDS copies (ASYNCcnt) so K+32 staging overlaps WMMAs.
// A fragments come straight from global (L2-resident, 2x reuse only).
//
// Fragment layout per CDNA5 ISA (16-bit operands, wave32):
//   lanes 0-15  : rows, VGPR0-3 = K 0..7 (pairs), VGPR4-7 = K 16..23
//   lanes 16-31 : rows, VGPR0-3 = K 8..15,        VGPR4-7 = K 24..31
// ---------------------------------------------------------------------------
__device__ __forceinline__ v16bf load_frag(const u16* p) {
  union U { v16bf v; v8bf h[2]; } u;
  u.h[0] = *(const v8bf*)(p);        // K base .. base+7
  u.h[1] = *(const v8bf*)(p + 16);   // K base+16 .. base+23
  return u.v;
}

#define WMMA_BF16(A, B, C) \
  __builtin_amdgcn_wmma_f32_16x16x32_bf16(false, (A), false, (B), (short)0, (C), false, false)

__global__ __launch_bounds__(256) void gemm_xwT_bf16x3(
    const u16* __restrict__ Ahi, const u16* __restrict__ Alo,
    const u16* __restrict__ Whi, const u16* __restrict__ Wlo,
    float* __restrict__ C, int M, int N, int K) {
  __shared__ __align__(16) u16 wtile[2][2][128][32];  // [buf][hi/lo][row][k] = 32 KB

  const int tid  = threadIdx.x;
  const int lane = tid & 31;
  const int wave = tid >> 5;
  const int mw   = wave >> 1;                  // 0..3  (M position)
  const int g    = wave & 1;                   // 0..1  (N position)
  const int m0   = blockIdx.y * 128 + mw * 32;
  const int nblk = blockIdx.x * 128;
  const int lr   = lane & 15;
  const int koff = (lane >> 4) << 3;           // +8 elems for upper half-wave

  // --- cooperative W-tile staging: 1024 chunks of 16B, 4 per thread ---
  auto copy_tile = [&](int buf, int k) {
#pragma unroll
    for (int c = 0; c < 4; c++) {
      const int chunk = tid + 256 * c;         // 0..1023
      const int mat   = chunk >> 9;            // 0 = hi, 1 = lo
      const int r     = (chunk >> 2) & 127;    // W row within tile
      const int cp    = chunk & 3;             // 8-elem chunk within K-slab
      const u16* src  = (mat ? Wlo : Whi) + (size_t)(nblk + r) * K + k + cp * 8;
      u16* dst        = &wtile[buf][mat][r][cp * 8];
#if HAVE_ASYNC_LDS
      __builtin_amdgcn_global_load_async_to_lds_b128(
          (AS1 v4i*)src, (AS3 v4i*)dst, 0, 0);
#else
      *(v8bf*)dst = *(const v8bf*)src;
#endif
    }
  };

  auto lds_frag = [&](int buf, int mat, int row) -> v16bf {
    union U { v16bf v; v8bf h[2]; } u;
    const u16* p = &wtile[buf][mat][row][koff];
    u.h[0] = *(const v8bf*)(p);
    u.h[1] = *(const v8bf*)(p + 16);
    return u.v;
  };

  const u16* aHi0 = Ahi + (size_t)(m0 + lr) * K + koff;
  const u16* aLo0 = Alo + (size_t)(m0 + lr) * K + koff;
  const u16* aHi1 = Ahi + (size_t)(m0 + 16 + lr) * K + koff;
  const u16* aLo1 = Alo + (size_t)(m0 + 16 + lr) * K + koff;

  v8f acc[2][4] = {};

  copy_tile(0, 0);
  for (int k = 0; k < K; k += 32) {
    const int cur = (k >> 5) & 1;
    if (k + 32 < K) {
      copy_tile(cur ^ 1, k + 32);  // prefetch next slab into other buffer
      async_wait_le4();            // older batch (current buffer) has landed
    } else {
      async_wait_le0();
    }
    __syncthreads();               // all waves' copies for buf[cur] visible

    v16bf ah0 = load_frag(aHi0 + k);
    v16bf al0 = load_frag(aLo0 + k);
    v16bf ah1 = load_frag(aHi1 + k);
    v16bf al1 = load_frag(aLo1 + k);

#pragma unroll
    for (int t = 0; t < 4; t++) {
      const int row = g * 64 + t * 16 + lr;
      v16bf bh = lds_frag(cur, 0, row);
      v16bf bl = lds_frag(cur, 1, row);
      acc[0][t] = WMMA_BF16(ah0, bh, acc[0][t]);  // hi*hi
      acc[0][t] = WMMA_BF16(ah0, bl, acc[0][t]);  // hi*lo
      acc[0][t] = WMMA_BF16(al0, bh, acc[0][t]);  // lo*hi
      acc[1][t] = WMMA_BF16(ah1, bh, acc[1][t]);
      acc[1][t] = WMMA_BF16(ah1, bl, acc[1][t]);
      acc[1][t] = WMMA_BF16(al1, bh, acc[1][t]);
    }
    __syncthreads();               // buf[cur] free for next iteration's copy
  }

  // C/D layout: VGPR r -> row base + r (+8 for lanes 16-31), lane&15 -> col.
#pragma unroll
  for (int ms = 0; ms < 2; ms++) {
    const int rbase = m0 + ms * 16 + ((lane >> 4) << 3);
#pragma unroll
    for (int t = 0; t < 4; t++) {
      const int col = nblk + g * 64 + t * 16 + lr;
#pragma unroll
      for (int r = 0; r < 8; r++) {
        C[(size_t)(rbase + r) * N + col] = acc[ms][t][r];
      }
    }
  }
}

// ---------------------------------------------------------------------------
// Fused LayerNorm + LSTM gate kernel. One block (256 thr = 8 wave32) per row.
// ---------------------------------------------------------------------------
__device__ __forceinline__ float wave_sum32(float v) {
#pragma unroll
  for (int o = 16; o > 0; o >>= 1) v += __shfl_down(v, o, 32);
  return v;
}
__device__ __forceinline__ float sigmoidf(float x) { return 1.0f / (1.0f + __expf(-x)); }

__global__ __launch_bounds__(256) void lncell_kernel(
    const float* __restrict__ Ag,   // [B,4096] = x @ W_ih^T
    const float* __restrict__ Hg,   // [B,4096] = hx @ W_hh^T
    const float* __restrict__ cx,   // [B,1024]
    const float* __restrict__ gi, const float* __restrict__ bi,
    const float* __restrict__ gh, const float* __restrict__ bhv,
    const float* __restrict__ gc, const float* __restrict__ bcv,
    float* __restrict__ hy, float* __restrict__ cy) {
  const int row  = blockIdx.x;
  const int tid  = threadIdx.x;
  const int lane = tid & 31;
  const int wave = tid >> 5;
  const float* Ar  = Ag + (size_t)row * 4096;
  const float* Hr  = Hg + (size_t)row * 4096;
  const float* cxr = cx + (size_t)row * 1024;

  __shared__ float red[8][4];

  // ---- stats over 4096-wide igate / hgate rows ----
  float sa = 0.f, sa2 = 0.f, sb = 0.f, sb2 = 0.f;
  for (int i = tid; i < 4096; i += 256) {
    float a = Ar[i]; sa += a; sa2 += a * a;
    float b = Hr[i]; sb += b; sb2 += b * b;
  }
  sa = wave_sum32(sa); sa2 = wave_sum32(sa2);
  sb = wave_sum32(sb); sb2 = wave_sum32(sb2);
  if (lane == 0) { red[wave][0] = sa; red[wave][1] = sa2; red[wave][2] = sb; red[wave][3] = sb2; }
  __syncthreads();
  float tA = 0.f, tA2 = 0.f, tB = 0.f, tB2 = 0.f;
#pragma unroll
  for (int w = 0; w < 8; w++) { tA += red[w][0]; tA2 += red[w][1]; tB += red[w][2]; tB2 += red[w][3]; }
  const float muA = tA * (1.0f / 4096.0f);
  const float rsA = rsqrtf(tA2 * (1.0f / 4096.0f) - muA * muA + 1e-5f);
  const float muB = tB * (1.0f / 4096.0f);
  const float rsB = rsqrtf(tB2 * (1.0f / 4096.0f) - muB * muB + 1e-5f);
  __syncthreads();  // red reused below

  // ---- gates + cell update ----
#define GATE(n) ((Ar[(n)] - muA) * rsA * gi[(n)] + bi[(n)] + \
                 (Hr[(n)] - muB) * rsB * gh[(n)] + bhv[(n)])
  float c4[4], o4[4];
  float sc = 0.f, sc2 = 0.f;
#pragma unroll
  for (int q = 0; q < 4; q++) {
    const int j = tid + q * 256;
    float iv = sigmoidf(GATE(j));
    float fv = sigmoidf(GATE(j + 1024));
    float gv = tanhf(GATE(j + 2048));
    float ov = sigmoidf(GATE(j + 3072));
    float c  = fv * cxr[j] + iv * gv;
    c4[q] = c; o4[q] = ov;
    sc += c; sc2 += c * c;
  }
#undef GATE

  // ---- LN over cell state (1024 wide) ----
  sc = wave_sum32(sc); sc2 = wave_sum32(sc2);
  if (lane == 0) { red[wave][0] = sc; red[wave][1] = sc2; }
  __syncthreads();
  float tC = 0.f, tC2 = 0.f;
#pragma unroll
  for (int w = 0; w < 8; w++) { tC += red[w][0]; tC2 += red[w][1]; }
  const float muC = tC * (1.0f / 1024.0f);
  const float rsC = rsqrtf(tC2 * (1.0f / 1024.0f) - muC * muC + 1e-5f);

#pragma unroll
  for (int q = 0; q < 4; q++) {
    const int j = tid + q * 256;
    float cyv = (c4[q] - muC) * rsC * gc[j] + bcv[j];
    cy[(size_t)row * 1024 + j] = cyv;
    hy[(size_t)row * 1024 + j] = o4[q] * tanhf(cyv);
  }
}

// ---------------------------------------------------------------------------
// Launch
// ---------------------------------------------------------------------------
extern "C" void kernel_launch(void* const* d_in, const int* in_sizes, int n_in,
                              void* d_out, int out_size, void* d_ws, size_t ws_size,
                              hipStream_t stream) {
  const int I = 1024, H = 1024;
  const int K = 1024, N = 4 * H;            // 4096
  const int B = in_sizes[0] / I;            // 8192

  const float* x   = (const float*)d_in[0];
  const float* hx  = (const float*)d_in[1];
  const float* cx  = (const float*)d_in[2];
  const float* Wih = (const float*)d_in[3];
  const float* Whh = (const float*)d_in[4];
  const float* gi  = (const float*)d_in[5];
  const float* bi  = (const float*)d_in[6];
  const float* gh  = (const float*)d_in[7];
  const float* bhv = (const float*)d_in[8];
  const float* gc  = (const float*)d_in[9];
  const float* bcv = (const float*)d_in[10];

  // Workspace layout (~352 MB total):
  char* ws = (char*)d_ws;
  size_t off = 0;
  float* Ag = (float*)(ws + off); off += (size_t)B * N * sizeof(float);   // 134 MB
  float* Hg = (float*)(ws + off); off += (size_t)B * N * sizeof(float);   // 134 MB
  u16* xhi = (u16*)(ws + off); off += (size_t)B * K * sizeof(u16);
  u16* xlo = (u16*)(ws + off); off += (size_t)B * K * sizeof(u16);
  u16* hhi = (u16*)(ws + off); off += (size_t)B * K * sizeof(u16);
  u16* hlo = (u16*)(ws + off); off += (size_t)B * K * sizeof(u16);
  u16* wih_hi = (u16*)(ws + off); off += (size_t)N * K * sizeof(u16);
  u16* wih_lo = (u16*)(ws + off); off += (size_t)N * K * sizeof(u16);
  u16* whh_hi = (u16*)(ws + off); off += (size_t)N * K * sizeof(u16);
  u16* whh_lo = (u16*)(ws + off); off += (size_t)N * K * sizeof(u16);

  const int nX = B * K;   // 8.4M
  const int nW = N * K;   // 4.2M
  split_kernel<<<(nX + 255) / 256, 256, 0, stream>>>(x,   xhi,    xlo,    nX);
  split_kernel<<<(nX + 255) / 256, 256, 0, stream>>>(hx,  hhi,    hlo,    nX);
  split_kernel<<<(nW + 255) / 256, 256, 0, stream>>>(Wih, wih_hi, wih_lo, nW);
  split_kernel<<<(nW + 255) / 256, 256, 0, stream>>>(Whh, whh_hi, whh_lo, nW);

  dim3 grid(N / 128, B / 128);  // (32, 64)
  gemm_xwT_bf16x3<<<grid, 256, 0, stream>>>(xhi, xlo, wih_hi, wih_lo, Ag, B, N, K);
  gemm_xwT_bf16x3<<<grid, 256, 0, stream>>>(hhi, hlo, whh_hi, whh_lo, Hg, B, N, K);

  float* hy = (float*)d_out;
  float* cy = hy + (size_t)B * H;
  lncell_kernel<<<B, 256, 0, stream>>>(Ag, Hg, cx, gi, bi, gh, bhv, gc, bcv, hy, cy);
}
